// GraphNetAutoCenter_19481971655235
// MI455X (gfx1250) — compile-verified, matrix operand order checked by reading, not run on hardware
//
#include <hip/hip_runtime.h>
#include <hip/hip_bf16.h>

typedef __attribute__((ext_vector_type(16))) _Float16 v16h;
typedef __attribute__((ext_vector_type(8)))  _Float16 v8h;
typedef __attribute__((ext_vector_type(4)))  _Float16 v4h;
typedef __attribute__((ext_vector_type(8)))  float    v8f;

// Build a 16x32 f16 WMMA operand fragment from an LDS row of 32 halfs.
// Per CDNA5 ISA 16-bit operand layout: lanes 0-15 hold K[0..7],K[16..23];
// lanes 16-31 hold K[8..15],K[24..31].  half = lane>>4.
__device__ __forceinline__ v16h make_frag(const _Float16* row, int half) {
    v8h c0 = *(const v8h*)(row + half * 8);
    v8h c1 = *(const v8h*)(row + 16 + half * 8);
    return __builtin_shufflevector(c0, c1, 0,1,2,3,4,5,6,7,8,9,10,11,12,13,14,15);
}

// Async global->LDS copy of 8 bytes per lane (CDNA5 GLOBAL_LOAD_ASYNC_TO_LDS_B64,
// tracked on ASYNCcnt).  lds_addr = wave-relative LDS byte offset = low 32 bits
// of the generic shared-aperture address (ISA 10.2).
__device__ __forceinline__ void async_copy_b64(unsigned lds_addr, const void* gaddr) {
    asm volatile("global_load_async_to_lds_b64 %0, %1, off"
                 :: "v"(lds_addr), "v"(gaddr) : "memory");
}
__device__ __forceinline__ void wait_async0() {
    asm volatile("s_wait_asynccnt 0x0" ::: "memory");
}
__device__ __forceinline__ unsigned lds_off(const void* p) {
    return (unsigned)(size_t)p;
}

// Fused  out = relu( A @ W' + b' )  with per-column sum / sum-of-squares
// accumulation for the subsequent batch-norm.  The previous layer's BN
// (scale s, shift t per input column) is folded into the weights:
//   relu((h*s+t) @ W + b) = relu(h @ (s (.) W) + (b + W^T t))
// so A staging is a pure copy/convert.  EDGE mode builds rows as
// concat(xh[src] (f16), coords[src]-newcoord[dst]).
// DIRECT mode requires kin_real == KINP (true for all call sites).
template<int KINP, int KOUT, bool EDGE, typename InT, typename OutT>
__global__ __launch_bounds__(256) void gemm_relu_stats(
    const InT* __restrict__ in, int in_stride,
    const float* __restrict__ in_scale, const float* __restrict__ in_shift,
    const _Float16* __restrict__ xh, const float* __restrict__ coords,
    const float* __restrict__ newcoord, const int* __restrict__ edges,
    const float* __restrict__ W, int w_ld, int kin_real, int kout_real,
    const float* __restrict__ bias,
    OutT* __restrict__ out, int out_stride,
    float* __restrict__ stat_sum, float* __restrict__ stat_sq,
    int nrows)
{
    constexpr int NTILES = KOUT / 16;
    constexpr int KTILES = KINP / 32;
    // Async staging used when A rows are raw f16 copies.
    constexpr bool ASYNC = EDGE || (sizeof(InT) == 2);

    __shared__ alignas(16) _Float16 Wt[KOUT][KINP];     // s (.) W^T, f16, padded
    __shared__ alignas(16) _Float16 At[2][128][32];     // double-buffered A tile
    __shared__ float s_bfold[KOUT];                     // W^T t  (bias fold)
    __shared__ float s_sum[KOUT];
    __shared__ float s_sq[KOUT];
    __shared__ int   s_src[128];
    __shared__ int   s_dst[128];

    const int tid  = threadIdx.x;
    const int wave = tid >> 5;
    const int lane = tid & 31;
    const int half = lane >> 4;
    const int l16  = lane & 15;
    const int rb   = blockIdx.x * 128;
    const bool has_ss = (in_scale != nullptr);

    // Stage scaled W^T into LDS as f16 (pad with zeros).
    for (int idx = tid; idx < KOUT * KINP; idx += 256) {
        int n = idx / KINP;
        int k = idx - n * KINP;
        float w = 0.f;
        if (k < kin_real && n < kout_real) {
            w = W[k * w_ld + n];
            if (has_ss) w *= in_scale[k];
        }
        Wt[n][k] = (_Float16)w;
    }
    // Bias fold: s_bfold[n] = sum_k t[k] * W[k][n]
    if (tid < KOUT) {
        float f = 0.f;
        if (has_ss && tid < kout_real) {
            for (int k = 0; k < kin_real; ++k)
                f += in_shift[k] * W[k * w_ld + tid];
        }
        s_bfold[tid] = f;
        s_sum[tid] = 0.f;
        s_sq[tid]  = 0.f;
    }
    if (EDGE && tid < 128) {
        int r = rb + tid;
        int e0 = 0, e1 = 0;
        if (r < nrows) { e0 = edges[2 * r]; e1 = edges[2 * r + 1]; }
        s_src[tid] = e0; s_dst[tid] = e1;
    }

    // A-tile staging into buffer `buf` (pure copy; BN folded into weights).
    auto stage = [&](int kt, int buf) {
        if (EDGE && kt == KTILES - 1) {
            // Tail tile: cols 128..130 hold coords[src]-newcoord[dst], rest 0.
            for (int idx = tid; idx < 128 * 32; idx += 256) {
                int r = idx >> 5;
                int k = idx & 31;
                float v = 0.f;
                if (k < 3) v = coords[s_src[r] * 3 + k] - newcoord[s_dst[r] * 3 + k];
                At[buf][r][k] = (_Float16)v;
            }
        } else if (ASYNC) {
            // Pure f16 copies: async global->LDS, 8 bytes per lane.
            // Padded rows read valid memory (src=0 / in-workspace); garbage
            // rows masked in the epilogue (WMMA rows are independent).
            #pragma unroll
            for (int it = 0; it < 4; ++it) {
                int idx = tid + it * 256;          // 0..1023
                int r   = idx >> 3;
                int c4  = (idx & 7) * 4;
                const _Float16* p = EDGE
                    ? (xh + s_src[r] * 128 + kt * 32 + c4)
                    : ((const _Float16*)in + (rb + r) * in_stride + kt * 32 + c4);
                async_copy_b64(lds_off(&At[buf][r][c4]), p);
            }
        } else {
            // f32 source: vector load + convert (guard: `in` may be a real
            // exact-size input buffer, so padded rows must not read OOB).
            #pragma unroll
            for (int it = 0; it < 4; ++it) {
                int idx = tid + it * 256;
                int r   = idx >> 3;
                int c4  = (idx & 7) * 4;
                int gr  = rb + r;
                float v0 = 0.f, v1 = 0.f, v2 = 0.f, v3 = 0.f;
                if (gr < nrows) {
                    float4 t = *(const float4*)((const float*)in + gr * in_stride + kt * 32 + c4);
                    v0 = t.x; v1 = t.y; v2 = t.z; v3 = t.w;
                }
                v4h o; o[0] = (_Float16)v0; o[1] = (_Float16)v1;
                       o[2] = (_Float16)v2; o[3] = (_Float16)v3;
                *(v4h*)&At[buf][r][c4] = o;
            }
        }
    };

    v8f acc[NTILES];
    #pragma unroll
    for (int t = 0; t < NTILES; ++t) acc[t] = v8f{};

    stage(0, 0);
    if (ASYNC) wait_async0();
    __syncthreads();

    for (int kt = 0; kt < KTILES; ++kt) {
        int cur = kt & 1;
        if (kt + 1 < KTILES) stage(kt + 1, cur ^ 1);   // overlaps WMMAs below
        v16h a = make_frag(&At[cur][wave * 16 + l16][0], half);
        #pragma unroll
        for (int t = 0; t < NTILES; ++t) {
            v16h b = make_frag(&Wt[t * 16 + l16][kt * 32], half);
            acc[t] = __builtin_amdgcn_wmma_f32_16x16x32_f16(
                false, a, false, b, (short)0, acc[t], false, false);
        }
        if (ASYNC) wait_async0();                      // tile kt+1 landed in LDS
        __syncthreads();
    }

    // Epilogue: (bias + fold) + relu, store, per-column stats.
    #pragma unroll
    for (int t = 0; t < NTILES; ++t) {
        int col = t * 16 + l16;
        float bcol = (col < kout_real) ? (bias[col] + s_bfold[col]) : 0.f;
        float psum = 0.f, psq = 0.f;
        #pragma unroll
        for (int j = 0; j < 8; ++j) {
            int row = rb + wave * 16 + half * 8 + j;
            float v = acc[t][j] + bcol;
            v = v > 0.f ? v : 0.f;
            if (row < nrows && col < kout_real) {
                out[row * out_stride + col] = (OutT)v;
                psum += v;
                psq  += v * v;
            }
        }
        atomicAdd(&s_sum[col], psum);
        atomicAdd(&s_sq[col],  psq);
    }
    __syncthreads();
    if (tid < KOUT && tid < kout_real) {
        atomicAdd(&stat_sum[tid], s_sum[tid]);
        atomicAdd(&stat_sq[tid],  s_sq[tid]);
    }
}

// mu/var -> per-column scale/shift:  y = h*scale + shift
__global__ void finalize_stats(const float* __restrict__ sum,
                               const float* __restrict__ sq,
                               const float* __restrict__ g,
                               const float* __restrict__ beta,
                               float* __restrict__ scale,
                               float* __restrict__ shift,
                               float inv_n, int ncols)
{
    int c = blockIdx.x * blockDim.x + threadIdx.x;
    if (c >= ncols) return;
    float mu  = sum[c] * inv_n;
    float var = sq[c] * inv_n - mu * mu;
    float rs  = rsqrtf(var + 1e-5f);
    float sc  = g[c] * rs;
    scale[c] = sc;
    shift[c] = beta[c] - mu * sc;
}

__global__ void zero_kernel(float* __restrict__ p, int n)
{
    int i = blockIdx.x * blockDim.x + threadIdx.x;
    if (i < n) p[i] = 0.f;
}

// f32 -> f16 copy of x (gather source; WMMA consumes f16 anyway)
__global__ void to_half_kernel(const float* __restrict__ in,
                               _Float16* __restrict__ out, int n4)
{
    int i = blockIdx.x * blockDim.x + threadIdx.x;
    if (i >= n4) return;
    float4 t = *(const float4*)(in + i * 4);
    v4h o; o[0] = (_Float16)t.x; o[1] = (_Float16)t.y;
           o[2] = (_Float16)t.z; o[3] = (_Float16)t.w;
    *(v4h*)(out + i * 4) = o;
}

// newcoord = coords + normalized offset (layer A2 output, stride 16, 3 used)
__global__ void newcoord_kernel(const float* __restrict__ coords,
                                const float* __restrict__ h,
                                const float* __restrict__ scale,
                                const float* __restrict__ shift,
                                float* __restrict__ newc, int N)
{
    int i = blockIdx.x * blockDim.x + threadIdx.x;
    if (i >= N * 3) return;
    int n = i / 3, c = i - n * 3;
    newc[i] = coords[i] + h[n * 16 + c] * scale[c] + shift[c];
}

// segment_max via int-bit atomicMax: agg init 0, result = max(0, max over
// edges).  Valid because positive-float ordering == int ordering and every
// negative-float bit pattern is a negative int (< bits of 0.0f).
__global__ void scatter_max_kernel(const _Float16* __restrict__ ef,
                                   const float* __restrict__ scale,
                                   const float* __restrict__ shift,
                                   const int* __restrict__ edges,
                                   int* __restrict__ agg, int total)
{
    int idx = blockIdx.x * blockDim.x + threadIdx.x;
    if (idx >= total) return;
    int e = idx >> 7;
    int c = idx & 127;
    float v = (float)ef[idx] * scale[c] + shift[c];
    int d = edges[2 * e + 1];
    atomicMax(agg + d * 128 + c, __float_as_int(v));
}

// out = normalized(h) + x
__global__ void final_add_kernel(const float* __restrict__ h,
                                 const float* __restrict__ scale,
                                 const float* __restrict__ shift,
                                 const float* __restrict__ x,
                                 float* __restrict__ out, int total)
{
    int idx = blockIdx.x * blockDim.x + threadIdx.x;
    if (idx >= total) return;
    int c = idx & 127;
    out[idx] = h[idx] * scale[c] + shift[c] + x[idx];
}

extern "C" void kernel_launch(void* const* d_in, const int* in_sizes, int n_in,
                              void* d_out, int out_size, void* d_ws, size_t ws_size,
                              hipStream_t stream)
{
    const float* x      = (const float*)d_in[0];
    const float* coords = (const float*)d_in[1];
    const int*   edges  = (const int*)d_in[3];
    const float* aW1  = (const float*)d_in[4];
    const float* ab1  = (const float*)d_in[5];
    const float* ag1  = (const float*)d_in[6];
    const float* abt1 = (const float*)d_in[7];
    const float* aW2  = (const float*)d_in[8];
    const float* ab2  = (const float*)d_in[9];
    const float* ag2  = (const float*)d_in[10];
    const float* abt2 = (const float*)d_in[11];
    const float* eW1  = (const float*)d_in[12];
    const float* eb1  = (const float*)d_in[13];
    const float* eg1  = (const float*)d_in[14];
    const float* ebt1 = (const float*)d_in[15];
    const float* eW2  = (const float*)d_in[16];
    const float* eb2  = (const float*)d_in[17];
    const float* eg2  = (const float*)d_in[18];
    const float* ebt2 = (const float*)d_in[19];
    const float* uW1  = (const float*)d_in[20];
    const float* ub1  = (const float*)d_in[21];
    const float* ug1  = (const float*)d_in[22];
    const float* ubt1 = (const float*)d_in[23];
    const float* uW2  = (const float*)d_in[24];
    const float* ub2  = (const float*)d_in[25];
    const float* ug2  = (const float*)d_in[26];
    const float* ubt2 = (const float*)d_in[27];

    const int N = in_sizes[0] / 128;
    const int K = in_sizes[3] / 2;

    // ---- workspace carving ----
    char* ws = (char*)d_ws;
    size_t off = 0;
    auto carve = [&](size_t bytes) -> char* {
        char* p = ws + off;
        off = (off + bytes + 255) & ~(size_t)255;
        return p;
    };
    float*     stats  = (float*)carve(6 * 512 * sizeof(float)); // [6][sum|sq|scale|shift][128]
    float*     buf_a1 = (float*)carve((size_t)N * 64  * 4);
    float*     buf_a2 = (float*)carve((size_t)N * 16  * 4);
    float*     newc   = (float*)carve((size_t)N * 3   * 4);
    _Float16*  xh     = (_Float16*)carve((size_t)N * 128 * 2);  // f16 copy of x
    _Float16*  ef     = (_Float16*)carve((size_t)K * 128 * 2);  // f16, reused in-place
    float*     agg    = (float*)carve((size_t)N * 128 * 4);
    float*     hu     = (float*)carve((size_t)N * 128 * 4);     // reused in-place

    auto SSUM = [&](int s) { return stats + s * 512; };
    auto SSQ  = [&](int s) { return stats + s * 512 + 128; };
    auto SSC  = [&](int s) { return stats + s * 512 + 256; };
    auto SSH  = [&](int s) { return stats + s * 512 + 384; };

    const int nbN = (N + 127) / 128;
    const int nbK = (K + 127) / 128;

    // init stats and agg (agg=0 is the segment_max(.,0) floor)
    zero_kernel<<<(6 * 512 + 255) / 256, 256, 0, stream>>>(stats, 6 * 512);
    zero_kernel<<<(N * 128 + 255) / 256, 256, 0, stream>>>(agg, N * 128);
    to_half_kernel<<<(N * 32 + 255) / 256, 256, 0, stream>>>(x, xh, N * 32);

    // ---- stage A: offset MLP on nodes ----
    gemm_relu_stats<128, 64, false, float, float><<<nbN, 256, 0, stream>>>(
        x, 128, nullptr, nullptr, nullptr, nullptr, nullptr, nullptr,
        aW1, 64, 128, 64, ab1, buf_a1, 64, SSUM(0), SSQ(0), N);
    finalize_stats<<<1, 128, 0, stream>>>(SSUM(0), SSQ(0), ag1, abt1, SSC(0), SSH(0), 1.f / N, 64);

    gemm_relu_stats<64, 16, false, float, float><<<nbN, 256, 0, stream>>>(
        buf_a1, 64, SSC(0), SSH(0), nullptr, nullptr, nullptr, nullptr,
        aW2, 3, 64, 3, ab2, buf_a2, 16, SSUM(1), SSQ(1), N);
    finalize_stats<<<1, 128, 0, stream>>>(SSUM(1), SSQ(1), ag2, abt2, SSC(1), SSH(1), 1.f / N, 3);

    newcoord_kernel<<<(N * 3 + 255) / 256, 256, 0, stream>>>(coords, buf_a2, SSC(1), SSH(1), newc, N);

    // ---- stage B: edge MLP (f16 intermediate + async LDS staging) ----
    gemm_relu_stats<160, 128, true, float, _Float16><<<nbK, 256, 0, stream>>>(
        nullptr, 0, nullptr, nullptr, xh, coords, newc, edges,
        eW1, 128, 131, 128, eb1, ef, 128, SSUM(2), SSQ(2), K);
    finalize_stats<<<1, 128, 0, stream>>>(SSUM(2), SSQ(2), eg1, ebt1, SSC(2), SSH(2), 1.f / K, 128);

    gemm_relu_stats<128, 128, false, _Float16, _Float16><<<nbK, 256, 0, stream>>>(
        ef, 128, SSC(2), SSH(2), nullptr, nullptr, nullptr, nullptr,
        eW2, 128, 128, 128, eb2, ef, 128, SSUM(3), SSQ(3), K);
    finalize_stats<<<1, 128, 0, stream>>>(SSUM(3), SSQ(3), eg2, ebt2, SSC(3), SSH(3), 1.f / K, 128);

    // ---- segment max ----
    scatter_max_kernel<<<(K * 128 + 255) / 256, 256, 0, stream>>>(
        ef, SSC(3), SSH(3), edges, (int*)agg, K * 128);

    // ---- stage C: update MLP on nodes ----
    gemm_relu_stats<128, 128, false, float, float><<<nbN, 256, 0, stream>>>(
        agg, 128, nullptr, nullptr, nullptr, nullptr, nullptr, nullptr,
        uW1, 128, 128, 128, ub1, hu, 128, SSUM(4), SSQ(4), N);
    finalize_stats<<<1, 128, 0, stream>>>(SSUM(4), SSQ(4), ug1, ubt1, SSC(4), SSH(4), 1.f / N, 128);

    gemm_relu_stats<128, 128, false, float, float><<<nbN, 256, 0, stream>>>(
        hu, 128, SSC(4), SSH(4), nullptr, nullptr, nullptr, nullptr,
        uW2, 128, 128, 128, ub2, hu, 128, SSUM(5), SSQ(5), N);
    finalize_stats<<<1, 128, 0, stream>>>(SSUM(5), SSQ(5), ug2, ubt2, SSC(5), SSH(5), 1.f / N, 128);

    final_add_kernel<<<(N * 128 + 255) / 256, 256, 0, stream>>>(
        hu, SSC(5), SSH(5), x, (float*)d_out, N * 128);

    (void)n_in; (void)out_size; (void)ws_size;
}
